// HistoryCrossAttention_74345883894510
// MI455X (gfx1250) — matrix operand order repaired and checked by reading, not compiled, not run on hardware
//
#include <hip/hip_runtime.h>

typedef __attribute__((ext_vector_type(8)))  __bf16 v8bf;
typedef __attribute__((ext_vector_type(16))) __bf16 v16bf;
typedef __attribute__((ext_vector_type(8)))  float  v8f;

union V16 { v16bf v; v8bf h[2]; };

constexpr int B_ = 8, T_ = 256, HL_ = 16, D_ = 1024, NH_ = 16, HD_ = 64, KV_ = 4096;
constexpr float LN_EPS_ = 1e-5f;

static __device__ __forceinline__ v8f wmma_bf16(v16bf a, v16bf b, v8f c) {
  return __builtin_amdgcn_wmma_f32_16x16x32_bf16(false, a, false, b, (short)0, c,
                                                 false, false);
}

// ---------------- weight transpose + f32->bf16 convert: dst[n][k] = src[k][n]
__global__ void wtrans_kernel(const float* __restrict__ src, __bf16* __restrict__ dst,
                              int K, int N) {
  long total = (long)K * N;
  for (long i = blockIdx.x * (long)blockDim.x + threadIdx.x; i < total;
       i += (long)gridDim.x * blockDim.x) {
    int n = (int)(i / K);
    int k = (int)(i % K);
    dst[i] = (__bf16)src[(long)k * N + n];
  }
}

// ---------------- Q = LN(x @ Wq + bq) per head, store bf16 (B,NH,T,HD)
// wave tile: 32 rows x 64 cols (one head)
__global__ __launch_bounds__(32)
void qproj_kernel(const float* __restrict__ x, const __bf16* __restrict__ WqT,
                  const float* __restrict__ bq, const float* __restrict__ gamma,
                  const float* __restrict__ beta, __bf16* __restrict__ qbuf) {
  const int lane = threadIdx.x, half = lane >> 4, l16 = lane & 15;
  const int mbase = blockIdx.x * 32;   // rows over B*T
  const int g = blockIdx.y;            // head (64-column group)
  const int kb = half * 16;
  const float* arow0 = x + (size_t)(mbase + l16) * D_;
  const float* arow1 = x + (size_t)(mbase + 16 + l16) * D_;

  v8f acc[2][4] = {};
  for (int kc = 0; kc < D_; kc += 32) {
    __builtin_prefetch(arow0 + kc + 32 + kb, 0, 3);
    __builtin_prefetch(arow1 + kc + 32 + kb, 0, 3);
    V16 a0, a1;
#pragma unroll
    for (int i = 0; i < 8; ++i) {
      a0.v[i]     = (__bf16)arow0[kc + kb + i];
      a0.v[i + 8] = (__bf16)arow0[kc + kb + 8 + i];
      a1.v[i]     = (__bf16)arow1[kc + kb + i];
      a1.v[i + 8] = (__bf16)arow1[kc + kb + 8 + i];
    }
#pragma unroll
    for (int t = 0; t < 4; ++t) {
      const __bf16* brow = WqT + (size_t)(g * HD_ + t * 16 + l16) * D_ + kc + half * 16;
      v16bf bf = *(const v16bf*)brow;
      acc[0][t] = wmma_bf16(a0.v, bf, acc[0][t]);
      acc[1][t] = wmma_bf16(a1.v, bf, acc[1][t]);
    }
  }
#pragma unroll
  for (int a2 = 0; a2 < 2; ++a2) {
    const int mb = mbase + a2 * 16;
#pragma unroll
    for (int t = 0; t < 4; ++t) {
      float bv = bq[g * HD_ + t * 16 + l16];
#pragma unroll
      for (int r = 0; r < 8; ++r) acc[a2][t][r] += bv;
    }
    // LayerNorm over 64 cols of each row (16 lanes of one half x 4 tiles)
    float mu[8], rs[8];
#pragma unroll
    for (int r = 0; r < 8; ++r) {
      float s = 0.f, q = 0.f;
#pragma unroll
      for (int t = 0; t < 4; ++t) { float v = acc[a2][t][r]; s += v; q += v * v; }
#pragma unroll
      for (int m = 1; m < 16; m <<= 1) { s += __shfl_xor(s, m, 32); q += __shfl_xor(q, m, 32); }
      float mean = s * (1.f / HD_);
      mu[r] = mean;
      rs[r] = rsqrtf(q * (1.f / HD_) - mean * mean + LN_EPS_);
    }
#pragma unroll
    for (int t = 0; t < 4; ++t) {
      float gm = gamma[t * 16 + l16], bt = beta[t * 16 + l16];
#pragma unroll
      for (int r = 0; r < 8; ++r) {
        float v = (acc[a2][t][r] - mu[r]) * rs[r] * gm + bt;
        int trow = mb + r + half * 8;
        int b = trow >> 8, tt = trow & (T_ - 1);
        qbuf[(((size_t)b * NH_ + g) * T_ + tt) * HD_ + t * 16 + l16] = (__bf16)v;
      }
    }
  }
}

// ---------------- KV = hist @ Wkv + bkv; K half gets LN -> kbuf(key,d);
// V half stored transposed -> vT(d,key). wave tile: 32 rows x 64 cols.
__global__ __launch_bounds__(32)
void kvproj_kernel(const float* __restrict__ hist, const __bf16* __restrict__ WkvT,
                   const float* __restrict__ bkv, const float* __restrict__ kgamma,
                   const float* __restrict__ kbeta, __bf16* __restrict__ kbuf,
                   __bf16* __restrict__ vT) {
  const int lane = threadIdx.x, half = lane >> 4, l16 = lane & 15;
  const int mbase = blockIdx.x * 32;   // rows over B*HL*T = 32768
  const int g = blockIdx.y;            // 0..31: 64-column group in 2*D
  const int kb = half * 16;
  const float* arow0 = hist + (size_t)(mbase + l16) * D_;
  const float* arow1 = hist + (size_t)(mbase + 16 + l16) * D_;

  v8f acc[2][4] = {};
  for (int kc = 0; kc < D_; kc += 32) {
    __builtin_prefetch(arow0 + kc + 32 + kb, 0, 3);
    __builtin_prefetch(arow1 + kc + 32 + kb, 0, 3);
    V16 a0, a1;
#pragma unroll
    for (int i = 0; i < 8; ++i) {
      a0.v[i]     = (__bf16)arow0[kc + kb + i];
      a0.v[i + 8] = (__bf16)arow0[kc + kb + 8 + i];
      a1.v[i]     = (__bf16)arow1[kc + kb + i];
      a1.v[i + 8] = (__bf16)arow1[kc + kb + 8 + i];
    }
#pragma unroll
    for (int t = 0; t < 4; ++t) {
      const __bf16* brow = WkvT + (size_t)(g * HD_ + t * 16 + l16) * D_ + kc + half * 16;
      v16bf bf = *(const v16bf*)brow;
      acc[0][t] = wmma_bf16(a0.v, bf, acc[0][t]);
      acc[1][t] = wmma_bf16(a1.v, bf, acc[1][t]);
    }
  }
#pragma unroll
  for (int a2 = 0; a2 < 2; ++a2)
#pragma unroll
    for (int t = 0; t < 4; ++t) {
      float bv = bkv[g * HD_ + t * 16 + l16];
#pragma unroll
      for (int r = 0; r < 8; ++r) acc[a2][t][r] += bv;
    }

  if (g < NH_) {  // K head: LayerNorm + row-major (key,d) store
#pragma unroll
    for (int a2 = 0; a2 < 2; ++a2) {
      const int mb = mbase + a2 * 16;
      float mu[8], rs[8];
#pragma unroll
      for (int r = 0; r < 8; ++r) {
        float s = 0.f, q = 0.f;
#pragma unroll
        for (int t = 0; t < 4; ++t) { float v = acc[a2][t][r]; s += v; q += v * v; }
#pragma unroll
        for (int m = 1; m < 16; m <<= 1) { s += __shfl_xor(s, m, 32); q += __shfl_xor(q, m, 32); }
        float mean = s * (1.f / HD_);
        mu[r] = mean;
        rs[r] = rsqrtf(q * (1.f / HD_) - mean * mean + LN_EPS_);
      }
#pragma unroll
      for (int t = 0; t < 4; ++t) {
        float gm = kgamma[t * 16 + l16], bt = kbeta[t * 16 + l16];
#pragma unroll
        for (int r = 0; r < 8; ++r) {
          float v = (acc[a2][t][r] - mu[r]) * rs[r] * gm + bt;
          int row = mb + r + half * 8;
          int bb = row >> 12, key = row & (KV_ - 1);
          kbuf[(((size_t)bb * NH_ + g) * KV_ + key) * HD_ + t * 16 + l16] = (__bf16)v;
        }
      }
    }
  } else {        // V head: transposed (d,key) store, 8 consecutive keys per lane
    const int h = g - NH_;
    const int bb = mbase >> 12;
#pragma unroll
    for (int a2 = 0; a2 < 2; ++a2) {
      const int k0 = ((mbase + a2 * 16) & (KV_ - 1)) + half * 8;
#pragma unroll
      for (int t = 0; t < 4; ++t) {
        v8bf pack;
#pragma unroll
        for (int r = 0; r < 8; ++r) pack[r] = (__bf16)acc[a2][t][r];
        int d = t * 16 + l16;
        *(v8bf*)(vT + (((size_t)bb * NH_ + h) * HD_ + d) * KV_ + k0) = pack;
      }
    }
  }
}

// ---------------- attention with SSMax (no max-subtraction, matches reference)
// wave tile: 32 queries; K/V fragments shared by both 16-query blocks
__global__ __launch_bounds__(32)
void attn_kernel(const __bf16* __restrict__ qbuf, const __bf16* __restrict__ kbuf,
                 const __bf16* __restrict__ vT, const int* __restrict__ hmask,
                 __bf16* __restrict__ obuf) {
  __shared__ __bf16 ptile[2][16 * 32];  // P tiles: C-layout -> A-layout bounce
  const int lane = threadIdx.x, half = lane >> 4, l16 = lane & 15;
  const int bh = blockIdx.x;            // b*NH + h
  const int b = bh >> 4, h = bh & 15;
  const int qt = blockIdx.y;            // 32-query tile (0..7)
  const int kb = half * 16;
  // exponent multiplier: SSMAX_S * ln(HL*T) * (1/sqrt(HD))
  constexpr float C2 = 0.43f * 8.317766166719343f * 0.125f;

  V16 qa[2][2];
#pragma unroll
  for (int a2 = 0; a2 < 2; ++a2) {
    const __bf16* qrow = qbuf + ((size_t)bh * T_ + qt * 32 + a2 * 16 + l16) * HD_;
#pragma unroll
    for (int c = 0; c < 2; ++c) {
      qa[a2][c].h[0] = *(const v8bf*)(qrow + c * 32 + kb);
      qa[a2][c].h[1] = *(const v8bf*)(qrow + c * 32 + kb + 8);
    }
  }
  const __bf16* kh = kbuf + (size_t)bh * KV_ * HD_;
  const __bf16* vh = vT + (size_t)bh * HD_ * KV_;

  v8f acc[2][4] = {};
  float dpart[2][8] = {};

  for (int hl = 0; hl < HL_; ++hl) {
    if (hmask[b * HL_ + hl] == 0) continue;   // uniform per wave: scalar branch
    for (int kt = 0; kt < T_ / 32; ++kt) {
      const int key0 = hl * T_ + kt * 32;
#pragma unroll
      for (int sub = 0; sub < 2; ++sub) {
        const __bf16* krow = kh + (size_t)(key0 + sub * 16 + l16) * HD_ + half * 16;
        v16bf k0f = *(const v16bf*)krow;
        v16bf k1f = *(const v16bf*)(krow + 32);
#pragma unroll
        for (int a2 = 0; a2 < 2; ++a2) {
          v8f s = {};
          s = wmma_bf16(qa[a2][0].v, k0f, s);
          s = wmma_bf16(qa[a2][1].v, k1f, s);
#pragma unroll
          for (int r = 0; r < 8; ++r) {
            float p = __expf(s[r] * C2);
            dpart[a2][r] += p;
            ptile[a2][(r + half * 8) * 32 + sub * 16 + l16] = (__bf16)p;
          }
        }
      }
      __syncthreads();
      V16 pa[2];
#pragma unroll
      for (int a2 = 0; a2 < 2; ++a2) {
        pa[a2].h[0] = *(const v8bf*)&ptile[a2][l16 * 32 + kb];
        pa[a2].h[1] = *(const v8bf*)&ptile[a2][l16 * 32 + kb + 8];
      }
#pragma unroll
      for (int dt = 0; dt < 4; ++dt) {
        const __bf16* vrow = vh + (size_t)(dt * 16 + l16) * KV_ + key0 + half * 16;
        v16bf vbf = *(const v16bf*)vrow;
        acc[0][dt] = wmma_bf16(pa[0].v, vbf, acc[0][dt]);
        acc[1][dt] = wmma_bf16(pa[1].v, vbf, acc[1][dt]);
      }
      __syncthreads();
    }
  }

#pragma unroll
  for (int a2 = 0; a2 < 2; ++a2) {
    float rden[8];
#pragma unroll
    for (int r = 0; r < 8; ++r) {
      float s = dpart[a2][r];
#pragma unroll
      for (int m = 1; m < 16; m <<= 1) s += __shfl_xor(s, m, 32);
      rden[r] = 1.f / (s + 1e-6f);
    }
#pragma unroll
    for (int dt = 0; dt < 4; ++dt)
#pragma unroll
      for (int r = 0; r < 8; ++r) {
        float v = acc[a2][dt][r] * rden[r];
        int trow = qt * 32 + a2 * 16 + r + half * 8;
        obuf[((size_t)b * T_ + trow) * D_ + h * HD_ + dt * 16 + l16] = (__bf16)v;
      }
  }
}

// ---------------- out = obuf @ Wout (fp32 result), 32x64 wave tile
__global__ __launch_bounds__(32)
void outproj_kernel(const __bf16* __restrict__ obuf, const __bf16* __restrict__ WoutT,
                    float* __restrict__ out) {
  const int lane = threadIdx.x, half = lane >> 4, l16 = lane & 15;
  const int mbase = blockIdx.x * 32;
  const int g = blockIdx.y;
  const int kb = half * 16;
  const __bf16* arow0 = obuf + (size_t)(mbase + l16) * D_;
  const __bf16* arow1 = obuf + (size_t)(mbase + 16 + l16) * D_;

  v8f acc[2][4] = {};
  for (int kc = 0; kc < D_; kc += 32) {
    V16 a0, a1;
    a0.h[0] = *(const v8bf*)(arow0 + kc + kb);
    a0.h[1] = *(const v8bf*)(arow0 + kc + kb + 8);
    a1.h[0] = *(const v8bf*)(arow1 + kc + kb);
    a1.h[1] = *(const v8bf*)(arow1 + kc + kb + 8);
#pragma unroll
    for (int t = 0; t < 4; ++t) {
      const __bf16* brow = WoutT + (size_t)(g * HD_ + t * 16 + l16) * D_ + kc + half * 16;
      v16bf bf = *(const v16bf*)brow;
      acc[0][t] = wmma_bf16(a0.v, bf, acc[0][t]);
      acc[1][t] = wmma_bf16(a1.v, bf, acc[1][t]);
    }
  }
#pragma unroll
  for (int a2 = 0; a2 < 2; ++a2)
#pragma unroll
    for (int t = 0; t < 4; ++t)
#pragma unroll
      for (int r = 0; r < 8; ++r)
        out[(size_t)(mbase + a2 * 16 + r + half * 8) * D_ + g * HD_ + t * 16 + l16] =
            acc[a2][t][r];
}

extern "C" void kernel_launch(void* const* d_in, const int* in_sizes, int n_in,
                              void* d_out, int out_size, void* d_ws, size_t ws_size,
                              hipStream_t stream) {
  const float* x     = (const float*)d_in[0];
  const float* hist  = (const float*)d_in[1];
  const int*   hm    = (const int*)d_in[2];
  const float* Wq    = (const float*)d_in[3];
  const float* bq    = (const float*)d_in[4];
  const float* Wkv   = (const float*)d_in[5];
  const float* bkv   = (const float*)d_in[6];
  const float* Wout  = (const float*)d_in[7];
  const float* qg    = (const float*)d_in[8];
  const float* qb    = (const float*)d_in[9];
  const float* kg    = (const float*)d_in[10];
  const float* kbeta = (const float*)d_in[11];
  float* out = (float*)d_out;

  char* ws = (char*)d_ws;
  size_t off = 0;
  auto take = [&](size_t bytes) {
    char* p = ws + off;
    off += (bytes + 255) & ~(size_t)255;
    return p;
  };
  __bf16* WqT   = (__bf16*)take((size_t)D_ * D_ * 2);
  __bf16* WkvT  = (__bf16*)take((size_t)2 * D_ * D_ * 2);
  __bf16* WoutT = (__bf16*)take((size_t)D_ * D_ * 2);
  __bf16* qbuf  = (__bf16*)take((size_t)B_ * NH_ * T_ * HD_ * 2);
  __bf16* kbuf  = (__bf16*)take((size_t)B_ * NH_ * KV_ * HD_ * 2);
  __bf16* vTb   = (__bf16*)take((size_t)B_ * NH_ * HD_ * KV_ * 2);
  __bf16* obuf  = (__bf16*)take((size_t)B_ * T_ * D_ * 2);
  (void)ws_size; (void)in_sizes; (void)n_in; (void)out_size;

  wtrans_kernel<<<2048, 256, 0, stream>>>(Wq, WqT, D_, D_);
  wtrans_kernel<<<4096, 256, 0, stream>>>(Wkv, WkvT, D_, 2 * D_);
  wtrans_kernel<<<2048, 256, 0, stream>>>(Wout, WoutT, D_, D_);
  qproj_kernel<<<dim3((B_ * T_) / 32, NH_), 32, 0, stream>>>(x, WqT, bq, qg, qb, qbuf);
  kvproj_kernel<<<dim3((B_ * HL_ * T_) / 32, 2 * NH_), 32, 0, stream>>>(
      hist, WkvT, bkv, kg, kbeta, kbuf, vTb);
  attn_kernel<<<dim3(B_ * NH_, T_ / 32), 32, 0, stream>>>(qbuf, kbuf, vTb, hm, obuf);
  outproj_kernel<<<dim3((B_ * T_) / 32, NH_), 32, 0, stream>>>(obuf, WoutT, out);
}
